// ParallelMoELinear_48275432407523
// MI455X (gfx1250) — matrix operand order repaired and checked by reading, not compiled, not run on hardware
//
#include <hip/hip_runtime.h>
#include <hip/hip_bf16.h>

typedef __attribute__((ext_vector_type(16))) _Float16 v16h;
typedef __attribute__((ext_vector_type(8)))  _Float16 v8h;
typedef __attribute__((ext_vector_type(8)))  float    v8f;
typedef __attribute__((ext_vector_type(4)))  float    v4f;

#define IN_F   1024
#define OUT_F  4096
#define BM     128
#define BN     128
#define BK     32
#define KP     40   // LDS row pitch in halves: 32 data + 8 pad; 80B = 16B multiple

union Frag { v16h v; v8h h[2]; };

__device__ __forceinline__ v16h ld_frag(const _Float16* row, int c0, int c1) {
    Frag u;
    u.h[0] = *(const v8h*)(row + c0);   // ds_load_b128
    u.h[1] = *(const v8h*)(row + c1);   // ds_load_b128
    return u.v;
}

// split x into f16 hi + f16 lo (exact residual; ~22 mantissa bits combined)
__device__ __forceinline__ void split16(const float* __restrict__ xv,
                                        v8h& h0, v8h& h1, v8h& l0, v8h& l1) {
    #pragma unroll
    for (int j = 0; j < 8; ++j) {
        _Float16 h = (_Float16)xv[j];
        h0[j] = h; l0[j] = (_Float16)(xv[j] - (float)h);
    }
    #pragma unroll
    for (int j = 0; j < 8; ++j) {
        _Float16 h = (_Float16)xv[8 + j];
        h1[j] = h; l1[j] = (_Float16)(xv[8 + j] - (float)h);
    }
}

__launch_bounds__(256)
__global__ void moe_ragged_gemm_f16x3(
        const float* __restrict__ A,      // [T, 1024]
        const float* __restrict__ W,      // [E, 1024, 4096]
        const float* __restrict__ bias,   // [E, 4096]
        const int*   __restrict__ gs,     // [E]
        float* __restrict__ out,          // [T, 4096]
        int numM, int NE)
{
    __shared__ __align__(16) _Float16 Ah[BM][KP];
    __shared__ __align__(16) _Float16 Al[BM][KP];
    __shared__ __align__(16) _Float16 Bh[BN][KP];   // transposed: [n][k]
    __shared__ __align__(16) _Float16 Bl[BN][KP];

    const int bid  = blockIdx.x;
    const int mblk = bid % numM;      // m fastest: concurrent blocks share a weight panel in L2
    const int nblk = bid / numM;
    const int m0   = mblk * BM;
    const int n0   = nblk * BN;

    // Ragged expert lookup. Tokens are sorted by expert; group sizes are all
    // multiples of 256, so a 128-row tile never straddles an expert boundary.
    int e = NE - 1, off = 0;
    for (int i = 0; i < NE; ++i) {
        int g = gs[i];
        if (m0 < off + g) { e = i; break; }
        off += g;
    }

    const int tid  = threadIdx.x;
    const int lane = tid & 31;
    const int wave = tid >> 5;
    const int wm   = wave & 3;        // 4 wave-rows x 32 rows
    const int wn   = wave >> 2;       // 2 wave-cols x 64 cols
    const int lmod = lane & 15;
    const int lhi  = lane >> 4;       // half-wave select

    // staging indices
    const int am  = tid >> 1;         // 0..127 (A row)
    const int ak  = (tid & 1) * 16;   // 0 or 16 (A k-half)
    const int bn  = tid & 127;        // 0..127 (B column -> LDS row)
    const int bkg = (tid >> 7) * 16;  // 0 or 16 (B k-half)

    const float* Abase = A + (size_t)m0 * IN_F + (size_t)am * IN_F + ak;
    const float* Wbase = W + (size_t)e * IN_F * OUT_F + n0
                         + (size_t)bkg * OUT_F + bn;

    v8f acc[2][4];
    #pragma unroll
    for (int i = 0; i < 2; ++i)
        #pragma unroll
        for (int j = 0; j < 4; ++j)
            #pragma unroll
            for (int r = 0; r < 8; ++r) acc[i][j][r] = 0.0f;

    // ---- register staging buffers (single buffer: stores precede next loads)
    __align__(16) float xvA[16];
    float xvB[16];

    // prologue: fetch K-tile 0 into registers
    {
        const float* sa = Abase;
        *(v4f*)&xvA[0]  = *(const v4f*)(sa);
        *(v4f*)&xvA[4]  = *(const v4f*)(sa + 4);
        *(v4f*)&xvA[8]  = *(const v4f*)(sa + 8);
        *(v4f*)&xvA[12] = *(const v4f*)(sa + 12);
        const float* sb = Wbase;
        #pragma unroll
        for (int i = 0; i < 16; ++i) xvB[i] = sb[(size_t)i * OUT_F];
    }

    for (int kk = 0; kk < IN_F; kk += BK) {
        __syncthreads();   // previous compute done reading LDS

        // ---- convert current registers -> hi/lo f16 tiles in LDS ----
        {
            v8h h0, h1, l0, l1;
            split16(xvA, h0, h1, l0, l1);
            *(v8h*)&Ah[am][ak]     = h0;
            *(v8h*)&Ah[am][ak + 8] = h1;
            *(v8h*)&Al[am][ak]     = l0;
            *(v8h*)&Al[am][ak + 8] = l1;
        }
        {
            v8h h0, h1, l0, l1;
            split16(xvB, h0, h1, l0, l1);
            *(v8h*)&Bh[bn][bkg]     = h0;
            *(v8h*)&Bh[bn][bkg + 8] = h1;
            *(v8h*)&Bl[bn][bkg]     = l0;
            *(v8h*)&Bl[bn][bkg + 8] = l1;
        }
        __syncthreads();

        // ---- prefetch next K-tile into registers; in flight during WMMAs ----
        const int kn = kk + BK;
        if (kn < IN_F) {
            const float* sa = Abase + kn;
            *(v4f*)&xvA[0]  = *(const v4f*)(sa);
            *(v4f*)&xvA[4]  = *(const v4f*)(sa + 4);
            *(v4f*)&xvA[8]  = *(const v4f*)(sa + 8);
            *(v4f*)&xvA[12] = *(const v4f*)(sa + 12);
            const float* sb = Wbase + (size_t)kn * OUT_F;
            #pragma unroll
            for (int i = 0; i < 16; ++i) xvB[i] = sb[(size_t)i * OUT_F];
        }

        // ---- compute: 24 WMMAs / wave / K-step ----
        // A frag (16-bit 16x32): lanes<16 hold K{0..7,16..23}; lanes>=16 K{8..15,24..31}
        v16h afh[2], afl[2];
        #pragma unroll
        for (int mf = 0; mf < 2; ++mf) {
            const _Float16* ra = &Ah[wm * 32 + mf * 16 + lmod][0];
            const _Float16* rl = &Al[wm * 32 + mf * 16 + lmod][0];
            afh[mf] = ld_frag(ra, lhi * 8, 16 + lhi * 8);
            afl[mf] = ld_frag(rl, lhi * 8, 16 + lhi * 8);
        }
        #pragma unroll
        for (int nf = 0; nf < 4; ++nf) {
            // B frag (32x16): lanes<16 hold K{0..7,16..23} of column N=lane; lanes>=16 K{8..15,24..31}
            const _Float16* rb  = &Bh[wn * 64 + nf * 16 + lmod][0];
            const _Float16* rbl = &Bl[wn * 64 + nf * 16 + lmod][0];
            v16h bfh = ld_frag(rb,  lhi * 8, 16 + lhi * 8);
            v16h bfl = ld_frag(rbl, lhi * 8, 16 + lhi * 8);
            #pragma unroll
            for (int mf = 0; mf < 2; ++mf) {
                acc[mf][nf] = __builtin_amdgcn_wmma_f32_16x16x32_f16(
                    false, afh[mf], false, bfh, (short)0, acc[mf][nf], false, false);
                acc[mf][nf] = __builtin_amdgcn_wmma_f32_16x16x32_f16(
                    false, afl[mf], false, bfh, (short)0, acc[mf][nf], false, false);
                acc[mf][nf] = __builtin_amdgcn_wmma_f32_16x16x32_f16(
                    false, afh[mf], false, bfl, (short)0, acc[mf][nf], false, false);
            }
        }
    }

    // ---- bias + store (C/D layout: VGPR r, lanes<16 -> M=r, N=lane; lanes>=16 -> M=8+r)
    // Output is write-once: non-temporal stores keep the 160MB fp32 working set in L2.
    #pragma unroll
    for (int nf = 0; nf < 4; ++nf) {
        const int n = n0 + wn * 64 + nf * 16 + lmod;
        const float bv = bias[(size_t)e * OUT_F + n];
        #pragma unroll
        for (int mf = 0; mf < 2; ++mf) {
            const int mrow = m0 + wm * 32 + mf * 16 + lhi * 8;
            float* o = out + (size_t)mrow * OUT_F + n;
            #pragma unroll
            for (int r = 0; r < 8; ++r)
                __builtin_nontemporal_store(acc[mf][nf][r] + bv, o + (size_t)r * OUT_F);
        }
    }
}

extern "C" void kernel_launch(void* const* d_in, const int* in_sizes, int n_in,
                              void* d_out, int out_size, void* d_ws, size_t ws_size,
                              hipStream_t stream) {
    const float* A    = (const float*)d_in[0];
    const float* W    = (const float*)d_in[1];
    const float* bias = (const float*)d_in[2];
    const int*   gs   = (const int*)d_in[3];
    float* out = (float*)d_out;

    const int T    = in_sizes[0] / IN_F;   // 8192
    const int NE   = in_sizes[3];          // 8
    const int numM = T / BM;               // 64
    const int numN = OUT_F / BN;           // 32

    dim3 grid(numM * numN), block(256);
    hipLaunchKernelGGL(moe_ragged_gemm_f16x3, grid, block, 0, stream,
                       A, W, bias, gs, out, numM, NE);
}
